// ArcticDecoderLayer_59047210385673
// MI455X (gfx1250) — matrix operand order repaired and checked by reading, not compiled, not run on hardware
//
#include <hip/hip_runtime.h>
#include <hip/hip_bf16.h>
#include <stdint.h>

typedef _Float16 v16h __attribute__((ext_vector_type(16)));
typedef float    v8f  __attribute__((ext_vector_type(8)));

#define S_LEN 1024
#define HDIM  2048
#define NH    32
#define HD    64
#define KVH   8
#define NEXP  8

__device__ __forceinline__ uint32_t lds_offset_u32(const void* p) {
    // HIP __shared__ pointers are flat; LDS aperture maps addr[31:0] -> LDS offset.
    return (uint32_t)(uintptr_t)p;
}

// ---------------------------------------------------------------------------
// WMMA GEMM: C[M,N](f32) = A_f16[M,K] @ B_f32[K,N] (B converted f32->f16)
// mode 0: C = acc ; mode 1: C = acc + res ; mode 2: C += scale[m*ss]*acc
// Block: 256 threads = 8 waves in 2(M) x 4(N); tile 32x128, K-step 32.
// A tile staged with global_load_async_to_lds_b64 (ASYNCcnt), B via b128 loads.
// ---------------------------------------------------------------------------
__global__ __launch_bounds__(256) void gemm_wmma_f16(
    const _Float16* __restrict__ A, const float* __restrict__ B,
    float* __restrict__ C, const float* __restrict__ res,
    const float* __restrict__ scale, int scale_stride,
    int M, int N, int K, int mode)
{
    __shared__ _Float16 lds_a[32][32];        // [m][k]
    __shared__ _Float16 lds_b[128][34];       // [n][k], padded (68B row stride)

    const int tid  = threadIdx.x;
    const int lane = tid & 31;
    const int wave = tid >> 5;
    const int wm   = wave & 1;
    const int wn   = wave >> 1;
    const int bm   = blockIdx.y * 32;
    const int bn   = blockIdx.x * 128;

    // per-thread async-copy coordinates for the A tile (32x64B = 256 x b64)
    const int arow = tid >> 3;                 // 0..31
    const int apart = tid & 7;                 // 0..7 (8 bytes each)
    const uint32_t a_lds = lds_offset_u32(&lds_a[0][0]) + arow * 64 + apart * 8;
    const _Float16* a_g0 = A + (size_t)(bm + arow) * K + apart * 4;

    v8f acc0 = {}, acc1 = {};

    for (int k0 = 0; k0 < K; k0 += 32) {
        // ---- async A tile: one b64 per lane, Memory -> LDS directly ----
        {
            uint64_t ga = (uint64_t)(uintptr_t)(a_g0 + k0);
            asm volatile("global_load_async_to_lds_b64 %0, %1, off"
                         :: "v"(a_lds), "v"(ga) : "memory");
        }
        // ---- B tile 32x128 f32, convert to f16, store K-transposed ----
        #pragma unroll
        for (int i = 0; i < 4; ++i) {
            int idx = tid + i * 256;           // 0..1023 float4 chunks
            int kk  = idx >> 5;                // 0..31
            int n4  = (idx & 31) * 4;          // 0..124
            float4 bv = *(const float4*)&B[(size_t)(k0 + kk) * N + bn + n4];
            lds_b[n4 + 0][kk] = (_Float16)bv.x;
            lds_b[n4 + 1][kk] = (_Float16)bv.y;
            lds_b[n4 + 2][kk] = (_Float16)bv.z;
            lds_b[n4 + 3][kk] = (_Float16)bv.w;
        }
        if (k0 + 32 < K)
            __builtin_prefetch(&B[(size_t)(k0 + 32) * N + bn], 0, 1);

        asm volatile("s_wait_asynccnt 0x0" ::: "memory");
        __syncthreads();

        // ---- fragments ----
        union { v16h v; uint32_t u[8]; } af, bf0, bf1;
        {
            int row = wm * 16 + (lane & 15);
            int kh  = (lane >> 4) * 16;
            #pragma unroll
            for (int p = 0; p < 8; ++p) {
                int kk = p * 2 + kh;
                af.u[p] = *(const uint32_t*)&lds_a[row][kk];
            }
        }
        {
            int col = wn * 32 + (lane & 15);
            int kh  = (lane >> 4) * 16;
            #pragma unroll
            for (int p = 0; p < 8; ++p) {
                bf0.u[p] = *(const uint32_t*)&lds_b[col][p * 2 + kh];
                bf1.u[p] = *(const uint32_t*)&lds_b[col + 16][p * 2 + kh];
            }
        }
        acc0 = __builtin_amdgcn_wmma_f32_16x16x32_f16(false, af.v, false, bf0.v,
                                                      (short)0, acc0, false, false);
        acc1 = __builtin_amdgcn_wmma_f32_16x16x32_f16(false, af.v, false, bf1.v,
                                                      (short)0, acc1, false, false);
        __syncthreads();
    }

    const int mbase = bm + wm * 16 + ((lane >> 4) << 3);
    #pragma unroll
    for (int t = 0; t < 2; ++t) {
        const int n = bn + wn * 32 + t * 16 + (lane & 15);
        #pragma unroll
        for (int r = 0; r < 8; ++r) {
            int m = mbase + r;
            size_t idx = (size_t)m * N + n;
            float v = (t == 0) ? acc0[r] : acc1[r];
            if (mode == 1) v += res[idx];
            else if (mode == 2) v = C[idx] + scale[(size_t)m * scale_stride] * v;
            C[idx] = v;
        }
    }
}

// ---------------------------------------------------------------------------
// RMSNorm (f32 in, f16 out), one block per row, H = HDIM (static)
// ---------------------------------------------------------------------------
__global__ __launch_bounds__(256) void rmsnorm_f16(
    const float* __restrict__ x, const float* __restrict__ w,
    _Float16* __restrict__ out)
{
    __shared__ float red[256];
    const int s = blockIdx.x;
    const float* xr = x + (size_t)s * HDIM;
    float acc = 0.f;
    #pragma unroll
    for (int i = 0; i < HDIM / 256; ++i) {
        float v = xr[threadIdx.x + i * 256];
        acc += v * v;
    }
    red[threadIdx.x] = acc;
    __syncthreads();
    for (int st = 128; st > 0; st >>= 1) {
        if (threadIdx.x < st) red[threadIdx.x] += red[threadIdx.x + st];
        __syncthreads();
    }
    float rs = rsqrtf(red[0] / (float)HDIM + 1e-6f);
    #pragma unroll
    for (int i = 0; i < HDIM / 256; ++i) {
        int j = threadIdx.x + i * 256;
        out[(size_t)s * HDIM + j] = (_Float16)(xr[j] * rs * w[j]);
    }
}

// ---------------------------------------------------------------------------
// RoPE + reshape to [head][S][HD] f16; v converted/reshaped only
// ---------------------------------------------------------------------------
__global__ __launch_bounds__(256) void rope_qkv(
    const float* __restrict__ qb, const float* __restrict__ kb,
    const float* __restrict__ vb, const int* __restrict__ pos,
    _Float16* __restrict__ qo, _Float16* __restrict__ ko, _Float16* __restrict__ vo)
{
    const int s = blockIdx.x;
    const float p = (float)pos[s];
    #pragma unroll
    for (int ii = 0; ii < (NH * HD) / 256; ++ii) {
        int i = threadIdx.x + ii * 256;
        int h = i >> 6, d = i & 63;
        int fi = d & 31;
        float inv = __powf(10000.f, -(2.f * fi) / 64.f);
        float sn, cs; __sincosf(p * inv, &sn, &cs);
        float xx = qb[(size_t)s * (NH * HD) + i];
        float xr = (d < 32) ? -qb[(size_t)s * (NH * HD) + h * 64 + d + 32]
                            :  qb[(size_t)s * (NH * HD) + h * 64 + d - 32];
        qo[((size_t)h * S_LEN + s) * HD + d] = (_Float16)(xx * cs + xr * sn);
    }
    #pragma unroll
    for (int ii = 0; ii < (KVH * HD) / 256; ++ii) {
        int i = threadIdx.x + ii * 256;
        int h = i >> 6, d = i & 63;
        int fi = d & 31;
        float inv = __powf(10000.f, -(2.f * fi) / 64.f);
        float sn, cs; __sincosf(p * inv, &sn, &cs);
        float xx = kb[(size_t)s * (KVH * HD) + i];
        float xr = (d < 32) ? -kb[(size_t)s * (KVH * HD) + h * 64 + d + 32]
                            :  kb[(size_t)s * (KVH * HD) + h * 64 + d - 32];
        ko[((size_t)h * S_LEN + s) * HD + d] = (_Float16)(xx * cs + xr * sn);
        vo[((size_t)h * S_LEN + s) * HD + d] = (_Float16)vb[(size_t)s * (KVH * HD) + i];
    }
}

// ---------------------------------------------------------------------------
// Flash-style causal GQA attention. One wave per (16-row q-tile, head).
// ---------------------------------------------------------------------------
__global__ __launch_bounds__(32) void attn_wmma(
    const _Float16* __restrict__ qf, const _Float16* __restrict__ kf,
    const _Float16* __restrict__ vf, const int* __restrict__ amask,
    _Float16* __restrict__ out)
{
    __shared__ _Float16 lds_p[16 * 32];
    const int lane = threadIdx.x & 31;
    const int qt   = blockIdx.x;
    const int head = blockIdx.y;
    const int kvh  = head >> 2;
    const int q0   = qt * 16;
    const int row  = lane & 15;
    const int half = lane >> 4;

    union U16 { v16h v; uint32_t u[8]; };

    U16 aq0, aq1;
    {
        const _Float16* qp = qf + ((size_t)head * S_LEN + q0 + row) * HD;
        int kh = half * 16;
        #pragma unroll
        for (int p = 0; p < 8; ++p) {
            int kk = p * 2 + kh;
            aq0.u[p] = *(const uint32_t*)(qp + kk);
            aq1.u[p] = *(const uint32_t*)(qp + 32 + kk);
        }
    }

    v8f o[4] = {{}, {}, {}, {}};
    float rmax[8], rsum[8];
    #pragma unroll
    for (int r = 0; r < 8; ++r) { rmax[r] = -1e30f; rsum[r] = 0.f; }

    const int nchunks = (q0 + 16 + 31) >> 5;
    for (int c = 0; c < nchunks; ++c) {
        const int kb = c << 5;
        v8f s[2];
        #pragma unroll
        for (int nt = 0; nt < 2; ++nt) {
            U16 b0, b1;
            const _Float16* kp = kf + ((size_t)kvh * S_LEN + kb + nt * 16 + row) * HD;
            int kh = half * 16;
            #pragma unroll
            for (int p = 0; p < 8; ++p) {
                b0.u[p] = *(const uint32_t*)(kp + p * 2 + kh);
                b1.u[p] = *(const uint32_t*)(kp + 32 + p * 2 + kh);
            }
            v8f z = {};
            z = __builtin_amdgcn_wmma_f32_16x16x32_f16(false, aq0.v, false, b0.v, (short)0, z, false, false);
            z = __builtin_amdgcn_wmma_f32_16x16x32_f16(false, aq1.v, false, b1.v, (short)0, z, false, false);
            s[nt] = z;
        }
        #pragma unroll
        for (int nt = 0; nt < 2; ++nt) {
            int kg  = kb + nt * 16 + row;
            int mok = amask[kg];
            #pragma unroll
            for (int r = 0; r < 8; ++r) {
                int qg = q0 + r + 8 * half;
                float v = s[nt][r] * 0.125f;
                if (kg > qg || mok == 0) v = -1e30f;
                s[nt][r] = v;
            }
        }
        float alpha[8];
        #pragma unroll
        for (int r = 0; r < 8; ++r) {
            float mx = fmaxf(s[0][r], s[1][r]);
            #pragma unroll
            for (int off = 8; off >= 1; off >>= 1)
                mx = fmaxf(mx, __shfl_xor(mx, off, 32));
            float nm = fmaxf(rmax[r], mx);
            float p0 = __expf(s[0][r] - nm);
            float p1 = __expf(s[1][r] - nm);
            s[0][r] = p0; s[1][r] = p1;
            float sm = p0 + p1;
            #pragma unroll
            for (int off = 8; off >= 1; off >>= 1)
                sm += __shfl_xor(sm, off, 32);
            alpha[r] = __expf(rmax[r] - nm);
            rsum[r]  = rsum[r] * alpha[r] + sm;
            rmax[r]  = nm;
        }
        #pragma unroll
        for (int nt = 0; nt < 4; ++nt)
            #pragma unroll
            for (int r = 0; r < 8; ++r) o[nt][r] *= alpha[r];

        #pragma unroll
        for (int nt = 0; nt < 2; ++nt)
            #pragma unroll
            for (int r = 0; r < 8; ++r)
                lds_p[(r + 8 * half) * 32 + nt * 16 + row] = (_Float16)s[nt][r];
        __syncthreads();
        U16 pf;
        {
            int kh = half * 16;
            #pragma unroll
            for (int p = 0; p < 8; ++p) {
                int kk = p * 2 + kh;
                pf.u[p] = *(const uint32_t*)&lds_p[row * 32 + kk];
            }
        }
        __syncthreads();

        const _Float16* vbase = vf + ((size_t)kvh * S_LEN + kb) * HD;
        #pragma unroll
        for (int nt = 0; nt < 4; ++nt) {
            U16 bv;
            int kh = half * 16;
            #pragma unroll
            for (int p = 0; p < 8; ++p) {
                int kk = p * 2 + kh;
                uint32_t lo = *(const uint16_t*)&vbase[(size_t)kk * HD + nt * 16 + row];
                uint32_t hi = *(const uint16_t*)&vbase[(size_t)(kk + 1) * HD + nt * 16 + row];
                bv.u[p] = lo | (hi << 16);
            }
            o[nt] = __builtin_amdgcn_wmma_f32_16x16x32_f16(false, pf.v, false, bv.v,
                                                           (short)0, o[nt], false, false);
        }
    }

    _Float16* op = out + (size_t)q0 * HDIM + head * HD;
    #pragma unroll
    for (int r = 0; r < 8; ++r) {
        int m = r + 8 * half;
        float inv = 1.0f / rsum[r];
        #pragma unroll
        for (int nt = 0; nt < 4; ++nt)
            op[(size_t)m * HDIM + nt * 16 + row] = (_Float16)(o[nt][r] * inv);
    }
}

// ---------------------------------------------------------------------------
// Gate: logits -> softmax -> top-2 renormalized dense routing weights
// ---------------------------------------------------------------------------
__global__ __launch_bounds__(256) void gate_topk(
    const float* __restrict__ x, const float* __restrict__ gw,
    float* __restrict__ wdense)
{
    __shared__ float red[256][NEXP];
    const int t = blockIdx.x;
    float acc[NEXP] = {};
    #pragma unroll
    for (int i = 0; i < HDIM / 256; ++i) {
        int h = threadIdx.x + i * 256;
        float xv = x[(size_t)t * HDIM + h];
        #pragma unroll
        for (int e = 0; e < NEXP; ++e) acc[e] += xv * gw[(size_t)h * NEXP + e];
    }
    #pragma unroll
    for (int e = 0; e < NEXP; ++e) red[threadIdx.x][e] = acc[e];
    __syncthreads();
    for (int st = 128; st > 0; st >>= 1) {
        if (threadIdx.x < st)
            #pragma unroll
            for (int e = 0; e < NEXP; ++e) red[threadIdx.x][e] += red[threadIdx.x + st][e];
        __syncthreads();
    }
    if (threadIdx.x == 0) {
        float lg[NEXP]; float mx = -1e30f;
        for (int e = 0; e < NEXP; ++e) { lg[e] = red[0][e]; mx = fmaxf(mx, lg[e]); }
        float sm = 0.f;
        for (int e = 0; e < NEXP; ++e) { lg[e] = __expf(lg[e] - mx); sm += lg[e]; }
        for (int e = 0; e < NEXP; ++e) lg[e] /= sm;
        int i0 = 0;
        for (int e = 1; e < NEXP; ++e) if (lg[e] > lg[i0]) i0 = e;
        int i1 = (i0 == 0) ? 1 : 0;
        for (int e = 0; e < NEXP; ++e) if (e != i0 && lg[e] > lg[i1]) i1 = e;
        float tot = lg[i0] + lg[i1];
        for (int e = 0; e < NEXP; ++e) wdense[(size_t)t * NEXP + e] = 0.f;
        wdense[(size_t)t * NEXP + i0] = lg[i0] / tot;
        wdense[(size_t)t * NEXP + i1] = lg[i1] / tot;
    }
}

__global__ void silu_mul_f16(const float* __restrict__ g1, const float* __restrict__ g3,
                             _Float16* __restrict__ up, int n)
{
    int i = blockIdx.x * blockDim.x + threadIdx.x;
    if (i < n) {
        float a = g1[i];
        up[i] = (_Float16)((a / (1.f + __expf(-a))) * g3[i]);
    }
}

__global__ void cvt_f16(const float* __restrict__ x, _Float16* __restrict__ y, int n)
{
    int i = blockIdx.x * blockDim.x + threadIdx.x;
    if (i < n) y[i] = (_Float16)x[i];
}

// ---------------------------------------------------------------------------
extern "C" void kernel_launch(void* const* d_in, const int* in_sizes, int n_in,
                              void* d_out, int out_size, void* d_ws, size_t ws_size,
                              hipStream_t stream)
{
    (void)in_sizes; (void)n_in; (void)out_size; (void)ws_size;
    const float* x        = (const float*)d_in[0];
    const int*   amask    = (const int*)d_in[1];
    const int*   pos      = (const int*)d_in[2];
    const float* wq       = (const float*)d_in[3];
    const float* wk       = (const float*)d_in[4];
    const float* wv       = (const float*)d_in[5];
    const float* wo       = (const float*)d_in[6];
    const float* norm1_w  = (const float*)d_in[7];
    const float* normr_w  = (const float*)d_in[8];
    const float* res_w1   = (const float*)d_in[9];
    const float* res_w3   = (const float*)d_in[10];
    const float* res_w2   = (const float*)d_in[11];
    const float* gate_w   = (const float*)d_in[12];
    const float* e_w1     = (const float*)d_in[13];
    const float* e_w3     = (const float*)d_in[14];
    const float* e_w2     = (const float*)d_in[15];
    float* out = (float*)d_out;

    const size_t MB = 1u << 20;
    char* ws = (char*)d_ws;
    _Float16* xf16  = (_Float16*)(ws + 0 * MB);   // 4 MB
    _Float16* hnorm = (_Float16*)(ws + 4 * MB);   // 4 MB
    float*    qbuf  = (float*)   (ws + 8 * MB);   // 8 MB
    float*    kbuf  = (float*)   (ws + 16 * MB);  // 2 MB
    float*    vbuf  = (float*)   (ws + 18 * MB);  // 2 MB
    _Float16* qf16  = (_Float16*)(ws + 20 * MB);  // 4 MB
    _Float16* kf16  = (_Float16*)(ws + 24 * MB);  // 1 MB
    _Float16* vf16  = (_Float16*)(ws + 25 * MB);  // 1 MB
    _Float16* attnb = (_Float16*)(ws + 26 * MB);  // 4 MB
    float*    h1    = (float*)   (ws + 30 * MB);  // 8 MB
    _Float16* hr    = (_Float16*)(ws + 38 * MB);  // 4 MB
    float*    g1    = (float*)   (ws + 42 * MB);  // 8 MB
    float*    g3    = (float*)   (ws + 50 * MB);  // 8 MB
    _Float16* up    = (_Float16*)(ws + 58 * MB);  // 4 MB
    float*    wdns  = (float*)   (ws + 62 * MB);  // 32 KB

    const int TOK = S_LEN * HDIM;
    const dim3 gN2048(HDIM / 128, S_LEN / 32);
    const dim3 gN512 (512 / 128,  S_LEN / 32);

    cvt_f16<<<(TOK + 255) / 256, 256, 0, stream>>>(x, xf16, TOK);
    rmsnorm_f16<<<S_LEN, 256, 0, stream>>>(x, norm1_w, hnorm);

    gemm_wmma_f16<<<gN2048, 256, 0, stream>>>(hnorm, wq, qbuf, nullptr, nullptr, 0,
                                              S_LEN, HDIM, HDIM, 0);
    gemm_wmma_f16<<<gN512, 256, 0, stream>>>(hnorm, wk, kbuf, nullptr, nullptr, 0,
                                             S_LEN, KVH * HD, HDIM, 0);
    gemm_wmma_f16<<<gN512, 256, 0, stream>>>(hnorm, wv, vbuf, nullptr, nullptr, 0,
                                             S_LEN, KVH * HD, HDIM, 0);
    rope_qkv<<<S_LEN, 256, 0, stream>>>(qbuf, kbuf, vbuf, pos, qf16, kf16, vf16);

    attn_wmma<<<dim3(S_LEN / 16, NH), 32, 0, stream>>>(qf16, kf16, vf16, amask, attnb);
    gemm_wmma_f16<<<gN2048, 256, 0, stream>>>(attnb, wo, h1, x, nullptr, 0,
                                              S_LEN, HDIM, HDIM, 1);

    rmsnorm_f16<<<S_LEN, 256, 0, stream>>>(h1, normr_w, hr);
    gemm_wmma_f16<<<gN2048, 256, 0, stream>>>(hr, res_w1, g1, nullptr, nullptr, 0,
                                              S_LEN, HDIM, HDIM, 0);
    gemm_wmma_f16<<<gN2048, 256, 0, stream>>>(hr, res_w3, g3, nullptr, nullptr, 0,
                                              S_LEN, HDIM, HDIM, 0);
    silu_mul_f16<<<(TOK + 255) / 256, 256, 0, stream>>>(g1, g3, up, TOK);
    gemm_wmma_f16<<<gN2048, 256, 0, stream>>>(up, res_w2, out, h1, nullptr, 0,
                                              S_LEN, HDIM, HDIM, 1);

    gate_topk<<<S_LEN, 256, 0, stream>>>(x, gate_w, wdns);
    for (int e = 0; e < NEXP; ++e) {
        const float* w1e = e_w1 + (size_t)e * HDIM * HDIM;
        const float* w3e = e_w3 + (size_t)e * HDIM * HDIM;
        const float* w2e = e_w2 + (size_t)e * HDIM * HDIM;
        gemm_wmma_f16<<<gN2048, 256, 0, stream>>>(xf16, w1e, g1, nullptr, nullptr, 0,
                                                  S_LEN, HDIM, HDIM, 0);
        gemm_wmma_f16<<<gN2048, 256, 0, stream>>>(xf16, w3e, g3, nullptr, nullptr, 0,
                                                  S_LEN, HDIM, HDIM, 0);
        silu_mul_f16<<<(TOK + 255) / 256, 256, 0, stream>>>(g1, g3, up, TOK);
        gemm_wmma_f16<<<gN2048, 256, 0, stream>>>(up, w2e, out, nullptr, wdns + e, NEXP,
                                                  S_LEN, HDIM, HDIM, 2);
    }
}
